// Top2Router_85126251807533
// MI455X (gfx1250) — matrix operand order repaired and checked by reading, not compiled, not run on hardware
//
#include <hip/hip_runtime.h>
#include <math.h>

#define N_EMBD     4096
#define N_EXPERTS  64
#define ROWS_TOTAL 16384            // 4 * 4096
#define KSLICE     512              // K columns handled per workgroup
#define NKSLICES   (N_EMBD / KSLICE)      // 8
#define ROWS_PER_WG 128             // 8 waves * 16 rows
#define NMGROUPS   (ROWS_TOTAL / ROWS_PER_WG)  // 128
#define NCHUNK     (KSLICE / 32)    // 16 K-chunks of 32
#define SLAB_ELEMS ((size_t)ROWS_TOTAL * N_EXPERTS)   // 1,048,576 floats / slab

typedef __attribute__((ext_vector_type(16))) __bf16   v16bf;
typedef __attribute__((ext_vector_type(8)))  float    v8f;
typedef __attribute__((ext_vector_type(8)))  unsigned v8u;

// v_perm_b32 selector: dst = {b.b3, b.b2, a.b3, a.b2}  (LSB..MSB: a2,a3,b2,b3)
// __builtin_amdgcn_perm(S0, S1, sel): sel bytes 0-3 pick from S1, 4-7 from S0.
#define BF16_PAIR_SEL 0x07060302u

// Pack two fp32 -> one dword of two bf16 "hi" parts: single v_perm_b32.
__device__ __forceinline__ unsigned pack_hi(float a, float b) {
    return __builtin_amdgcn_perm(__builtin_bit_cast(unsigned, b),
                                 __builtin_bit_cast(unsigned, a),
                                 BF16_PAIR_SEL);
}
// Pack the bf16 "lo" residuals (exact: x - bf16_trunc(x), then truncated).
__device__ __forceinline__ unsigned pack_lo(float a, float b) {
    unsigned ua = __builtin_bit_cast(unsigned, a);
    unsigned ub = __builtin_bit_cast(unsigned, b);
    float ra = a - __builtin_bit_cast(float, ua & 0xFFFF0000u);
    float rb = b - __builtin_bit_cast(float, ub & 0xFFFF0000u);
    return __builtin_amdgcn_perm(__builtin_bit_cast(unsigned, rb),
                                 __builtin_bit_cast(unsigned, ra),
                                 BF16_PAIR_SEL);
}

__device__ __forceinline__ v8f wmma_bf16(v16bf a, v16bf b, v8f c) {
    return __builtin_amdgcn_wmma_f32_16x16x32_bf16(
        /*neg_a=*/false, a, /*neg_b=*/false, b,
        /*c_mod=*/(short)0, c, /*reuse_a=*/false, /*reuse_b=*/false);
}

// ---------------------------------------------------------------------------
// Zero the logits accumulator (atomic-fallback path only).
// ---------------------------------------------------------------------------
__global__ void __launch_bounds__(256) zero_logits_kernel(float4* p) {
    p[(size_t)blockIdx.x * blockDim.x + threadIdx.x] =
        make_float4(0.f, 0.f, 0.f, 0.f);
}

// ---------------------------------------------------------------------------
// GEMM: logits += X[16384,4096] * W^T[4096,64] using split-bf16 WMMA.
// grid = (NMGROUPS, NKSLICES), block = 256 (8 waves). Wave w owns M-tile
// rows [mgrp*128 + w*16, +16) over K slice [blockIdx.y*512, +512).
// use_slabs != 0 : plain stores into slabs[blockIdx.y]; else atomicAdd.
// ---------------------------------------------------------------------------
__global__ void __launch_bounds__(256)
router_gemm_kernel(const float* __restrict__ X, const float* __restrict__ W,
                   float* __restrict__ out, int use_slabs) {
    extern __shared__ unsigned smem[];          // [2][16384] dwords = 128 KB
    unsigned* shi = smem;                       // hi-plane B fragments
    unsigned* slo = smem + 16384;               // lo-plane B fragments

    const int tid  = threadIdx.x;
    const int mgrp = blockIdx.x;
    const int ks   = blockIdx.y;
    const int K0   = ks * KSLICE;

    // ---- Stage W slice into LDS, split to bf16 hi/lo, pre-swizzled into the
    // WMMA B-matrix fragment layout: lane l holds expert n = nt*16 + (l&15),
    // K values (l>>4)*16 + e, e = 0..15 packed 2 per dword.
#pragma unroll
    for (int i = 0; i < 8; ++i) {
        int f  = tid + i * 256;                 // fragment-lane id, 2048 total
        int c  = f >> 7;                        // K-chunk 0..15
        int nt = (f >> 5) & 3;                  // N-tile 0..3
        int l  = f & 31;                        // lane slot
        int gg = l >> 4;
        int n  = nt * 16 + (l & 15);
        const float* src = W + (size_t)n * N_EMBD + K0 + c * 32 + gg * 16;
        float4 b0 = *(const float4*)(src + 0);
        float4 b1 = *(const float4*)(src + 4);
        float4 b2 = *(const float4*)(src + 8);
        float4 b3 = *(const float4*)(src + 12);
        v8u h, lo;
        h[0] = pack_hi(b0.x, b0.y);  lo[0] = pack_lo(b0.x, b0.y);
        h[1] = pack_hi(b0.z, b0.w);  lo[1] = pack_lo(b0.z, b0.w);
        h[2] = pack_hi(b1.x, b1.y);  lo[2] = pack_lo(b1.x, b1.y);
        h[3] = pack_hi(b1.z, b1.w);  lo[3] = pack_lo(b1.z, b1.w);
        h[4] = pack_hi(b2.x, b2.y);  lo[4] = pack_lo(b2.x, b2.y);
        h[5] = pack_hi(b2.z, b2.w);  lo[5] = pack_lo(b2.z, b2.w);
        h[6] = pack_hi(b3.x, b3.y);  lo[6] = pack_lo(b3.x, b3.y);
        h[7] = pack_hi(b3.z, b3.w);  lo[7] = pack_lo(b3.z, b3.w);
        *(v8u*)(shi + (size_t)f * 8) = h;       // 32B -> 2x ds_store_b128
        *(v8u*)(slo + (size_t)f * 8) = lo;
    }
    __syncthreads();

    const int wave = tid >> 5;
    const int lane = tid & 31;
    const int g    = lane >> 4;                 // half-wave group
    const int mrow = lane & 15;                 // row within M-tile (A layout)
    const int row0 = mgrp * ROWS_PER_WG + wave * 16;
    const float* xrow = X + (size_t)(row0 + mrow) * N_EMBD + K0;

    v8f acc[4] = {};                            // 16x64 fp32 accumulators

    for (int c = 0; c < NCHUNK; ++c) {
        // A fragment: row mrow; lanes 0-15 hold K 0-15, lanes 16-31
        // hold K 16-31 -> base offset g*16, contiguous (matches B).
        const float* p = xrow + c * 32 + g * 16;
        float4 a0 = *(const float4*)(p + 0);
        float4 a1 = *(const float4*)(p + 4);
        float4 a2 = *(const float4*)(p + 8);
        float4 a3 = *(const float4*)(p + 12);
        v8u ah, al;
        ah[0] = pack_hi(a0.x, a0.y);  al[0] = pack_lo(a0.x, a0.y);
        ah[1] = pack_hi(a0.z, a0.w);  al[1] = pack_lo(a0.z, a0.w);
        ah[2] = pack_hi(a1.x, a1.y);  al[2] = pack_lo(a1.x, a1.y);
        ah[3] = pack_hi(a1.z, a1.w);  al[3] = pack_lo(a1.z, a1.w);
        ah[4] = pack_hi(a2.x, a2.y);  al[4] = pack_lo(a2.x, a2.y);
        ah[5] = pack_hi(a2.z, a2.w);  al[5] = pack_lo(a2.z, a2.w);
        ah[6] = pack_hi(a3.x, a3.y);  al[6] = pack_lo(a3.x, a3.y);
        ah[7] = pack_hi(a3.z, a3.w);  al[7] = pack_lo(a3.z, a3.w);
        v16bf ahi = __builtin_bit_cast(v16bf, ah);
        v16bf alo = __builtin_bit_cast(v16bf, al);

#pragma unroll
        for (int nt = 0; nt < 4; ++nt) {
            const size_t foff = ((size_t)(c * 4 + nt) * 32 + lane) * 8;
            v16bf bhi = __builtin_bit_cast(v16bf, *(const v8u*)(shi + foff));
            v16bf blo = __builtin_bit_cast(v16bf, *(const v8u*)(slo + foff));
            // x*w ~= hi*hi + hi*lo + lo*hi   (fp32 accumulate)
            acc[nt] = wmma_bf16(ahi, bhi, acc[nt]);
            acc[nt] = wmma_bf16(ahi, blo, acc[nt]);
            acc[nt] = wmma_bf16(alo, bhi, acc[nt]);
        }
    }

    // C/D layout: VGPR r holds row (r + 8*g), column nt*16 + (lane&15).
    if (use_slabs) {
        float* slab = out + (size_t)ks * SLAB_ELEMS;
#pragma unroll
        for (int nt = 0; nt < 4; ++nt)
#pragma unroll
            for (int r = 0; r < 8; ++r) {
                int row = row0 + g * 8 + r;
                int col = nt * 16 + (lane & 15);
                slab[(size_t)row * N_EXPERTS + col] = acc[nt][r];
            }
    } else {
#pragma unroll
        for (int nt = 0; nt < 4; ++nt)
#pragma unroll
            for (int r = 0; r < 8; ++r) {
                int row = row0 + g * 8 + r;
                int col = nt * 16 + (lane & 15);
                atomicAdd(&out[(size_t)row * N_EXPERTS + col], acc[nt][r]);
            }
    }
}

// ---------------------------------------------------------------------------
// Top-2 + softmax over the top-2. One thread per row; sums nslabs partial
// slabs in fixed order (deterministic), keeps lowest index on ties like
// lax.top_k, values in descending order.
// ---------------------------------------------------------------------------
__global__ void __launch_bounds__(256)
top2_softmax_kernel(const float* __restrict__ slabs, int nslabs,
                    int* __restrict__ idx_out, float* __restrict__ gate_out) {
    int row = blockIdx.x * blockDim.x + threadIdx.x;
    if (row >= ROWS_TOTAL) return;

    // Accumulate the 64 logits for this row across slabs with float4 loads.
    float4 lg[16];
#pragma unroll
    for (int j = 0; j < 16; ++j) lg[j] = make_float4(0.f, 0.f, 0.f, 0.f);
    for (int s = 0; s < nslabs; ++s) {
        const float4* sp = (const float4*)(slabs + (size_t)s * SLAB_ELEMS +
                                           (size_t)row * N_EXPERTS);
#pragma unroll
        for (int j = 0; j < 16; ++j) {
            float4 t = sp[j];
            lg[j].x += t.x; lg[j].y += t.y; lg[j].z += t.z; lg[j].w += t.w;
        }
    }

    float v1 = -INFINITY, v2 = -INFINITY;
    int   i1 = 0,         i2 = 0;
#pragma unroll
    for (int j = 0; j < 16; ++j) {
        float vv[4] = { lg[j].x, lg[j].y, lg[j].z, lg[j].w };
#pragma unroll
        for (int q = 0; q < 4; ++q) {
            float v = vv[q];
            int   e = j * 4 + q;
            if (v > v1)      { v2 = v1; i2 = i1; v1 = v; i1 = e; }
            else if (v > v2) { v2 = v;  i2 = e; }
        }
    }
    // softmax over [v1, v2] with v1 >= v2
    float e1 = expf(v2 - v1);
    float s  = 1.f + e1;
    idx_out[row * 2 + 0]  = i1;
    idx_out[row * 2 + 1]  = i2;
    gate_out[row * 2 + 0] = 1.f / s;
    gate_out[row * 2 + 1] = e1 / s;
}

// ---------------------------------------------------------------------------
extern "C" void kernel_launch(void* const* d_in, const int* in_sizes, int n_in,
                              void* d_out, int out_size, void* d_ws, size_t ws_size,
                              hipStream_t stream) {
    const float* X = (const float*)d_in[0];   // [4, 4096, 4096] fp32
    const float* W = (const float*)d_in[1];   // [64, 4096] fp32

    int*   idx_out  = (int*)d_out;                       // [4,4096,2] int32
    float* gate_out = (float*)d_out + 2 * ROWS_TOTAL;    // [4,4096,2] fp32

    float* ws = (float*)d_ws;
    const size_t slab_bytes = SLAB_ELEMS * sizeof(float);

    dim3 gemm_grid(NMGROUPS, NKSLICES);
    const unsigned lds_bytes = 2u * 16384u * 4u;         // 128 KB

    if (ws_size >= (size_t)NKSLICES * slab_bytes) {
        // Deterministic path: one partial slab per K-slice (32 MB scratch).
        router_gemm_kernel<<<gemm_grid, 256, lds_bytes, stream>>>(
            X, W, ws, /*use_slabs=*/1);
        top2_softmax_kernel<<<ROWS_TOTAL / 256, 256, 0, stream>>>(
            ws, NKSLICES, idx_out, gate_out);
    } else {
        // Fallback: zero 4 MB accumulator then global_atomic_add_f32.
        zero_logits_kernel<<<(SLAB_ELEMS / 4) / 256, 256, 0, stream>>>(
            (float4*)ws);
        router_gemm_kernel<<<gemm_grid, 256, lds_bytes, stream>>>(
            X, W, ws, /*use_slabs=*/0);
        top2_softmax_kernel<<<ROWS_TOTAL / 256, 256, 0, stream>>>(
            ws, 1, idx_out, gate_out);
    }
}
